// CausalSelfAttention_48249662603393
// MI455X (gfx1250) — matrix operand order repaired and checked by reading, not compiled, not run on hardware
//
#include <hip/hip_runtime.h>
#include <hip/hip_bf16.h>

typedef __bf16 bf16_t;
typedef __bf16 v16bf __attribute__((ext_vector_type(16)));
typedef float  v8f   __attribute__((ext_vector_type(8)));

constexpr int kT    = 2048;
constexpr int kH    = 16;
constexpr int kHD   = 128;
constexpr int kQC   = kH * kHD;     // 2048 (q cols)
constexpr int kKC   = 4 * kHD;      // 512  (kv cols)
constexpr int kWIN  = 512;
constexpr int kRows = 2 * kT;       // 4096 = B*T
constexpr float kScale = 0.08838834764831845f; // 1/sqrt(128)

__device__ __forceinline__ v8f vzero8() {
  v8f v;
#pragma unroll
  for (int i = 0; i < 8; ++i) v[i] = 0.0f;
  return v;
}

__device__ __forceinline__ v8f wmma_bf16(v16bf a, v16bf b, v8f c) {
  // (neg_a, A, neg_b, B, c_mod, C, reuse_a, reuse_b)
  return __builtin_amdgcn_wmma_f32_16x16x32_bf16(false, a, false, b, (short)0, c,
                                                 false, false);
}

// ---------------------------------------------------------------- convert (x only)
__global__ __launch_bounds__(256) void cvt_f32_bf16(const float* __restrict__ in,
                                                    bf16_t* __restrict__ out,
                                                    size_t n) {
  size_t i = (size_t)blockIdx.x * blockDim.x + threadIdx.x;
  size_t stride = (size_t)gridDim.x * blockDim.x;
  for (; i < n; i += stride) out[i] = (bf16_t)in[i];
}

// ------------------------------------------------- fused convert + transpose weights
// in: R x C fp32 row-major   ->   out: C x R bf16 row-major (i.e. W^T)
__global__ __launch_bounds__(256) void cvt_transpose(const float* __restrict__ in,
                                                     bf16_t* __restrict__ out,
                                                     int R, int C) {
  __shared__ float tile[32][33];
  const int c0 = blockIdx.x * 32;
  const int r0 = blockIdx.y * 32;
  const int tx = threadIdx.x & 31;
  const int ty = threadIdx.x >> 5;         // 0..7
#pragma unroll
  for (int i = 0; i < 32; i += 8)
    tile[ty + i][tx] = in[(size_t)(r0 + ty + i) * C + c0 + tx];
  __syncthreads();
#pragma unroll
  for (int i = 0; i < 32; i += 8)
    out[(size_t)(c0 + ty + i) * R + r0 + tx] = (bf16_t)tile[tx][ty + i];
}

// ------------------------------------------------- transpose V: [b][t][kvh*128+d] -> [b*4+kvh][d][t]
__global__ __launch_bounds__(256) void transpose_v(const bf16_t* __restrict__ V,
                                                   bf16_t* __restrict__ Vt) {
  __shared__ bf16_t tile[32][33];
  const int slab = blockIdx.z;             // b*4 + kvh, 0..7
  const int b    = slab >> 2;
  const int kvh  = slab & 3;
  const bf16_t* src = V + (size_t)b * kT * kKC + (size_t)kvh * kHD; // row stride kKC
  bf16_t* dst = Vt + (size_t)slab * kHD * kT;                       // [128][T]
  const int d0 = blockIdx.x * 32;
  const int t0 = blockIdx.y * 32;
  const int tx = threadIdx.x & 31;
  const int ty = threadIdx.x >> 5;
#pragma unroll
  for (int i = 0; i < 32; i += 8)
    tile[ty + i][tx] = src[(size_t)(t0 + ty + i) * kKC + d0 + tx];
  __syncthreads();
#pragma unroll
  for (int i = 0; i < 32; i += 8)
    dst[(size_t)(d0 + ty + i) * kT + t0 + tx] = tile[tx][ty + i];
}

// ------------------------------------- RoPE (in place, bf16), optional output scale
__global__ __launch_bounds__(256) void rope_kernel(bf16_t* __restrict__ buf, int ncols,
                                                   float oscale) {
  const int row = blockIdx.x;              // 0..4095
  const int t = row & (kT - 1);            // position within sequence
  const int nwork = (ncols / kHD) * 64;    // heads * half-dim
  for (int idx = threadIdx.x; idx < nwork; idx += blockDim.x) {
    const int d  = idx & 63;
    const int hh = idx >> 6;
    const float inv = powf(10000.0f, -(float)(2 * d) * (1.0f / 128.0f));
    const float ang = (float)t * inv;
    const float c = cosf(ang), s = sinf(ang);
    const size_t base = (size_t)row * ncols + (size_t)hh * kHD + d;
    const float q0 = (float)buf[base];
    const float q1 = (float)buf[base + 64];
    buf[base]      = (bf16_t)((q0 * c - q1 * s) * oscale);
    buf[base + 64] = (bf16_t)((q1 * c + q0 * s) * oscale);
  }
}

// ---------------------------------------------------------------- tiled WMMA GEMM
// C[M,N] = A[M,K] * B[K,N] with B given TRANSPOSED (Bt = [N][K] bf16 row-major).
// Block = 8 waves; block computes a 16 x 512 strip of C. The 16x32 A k-slice is
// shared by all 8 waves and staged into LDS with async global->LDS copies
// (1 x b32 per lane), then each wave runs 4 WMMAs over its 64-col strip.
template <bool F32OUT>
__global__ __launch_bounds__(256) void gemm_bf16_wmma(const bf16_t* __restrict__ A,
                                                      const bf16_t* __restrict__ Bt,
                                                      void* __restrict__ Cc,
                                                      int M, int N, int K) {
  __shared__ bf16_t As[16 * 32];           // 1 KB A k-slice
  const int tid  = threadIdx.x;
  const int lane = tid & 31;
  // wave index made provably wave-uniform -> scalar addressing / SALU control flow
  const int wv   = __builtin_amdgcn_readfirstlane(tid >> 5);     // 0..7
  const int ngroups = N / 512;             // 512-wide column groups
  const int mt = (int)blockIdx.x / ngroups;
  const int ng = (int)blockIdx.x % ngroups;
  const int n0 = ng * 512 + wv * 64;       // this wave's 64-col strip

  const int hf    = lane >> 4;
  const int ln    = lane & 15;
  const int akoff = hf * 8;                // unused (A-frag also uses bkoff, halves of 16)
  const int bkoff = hf * 16;               // B-frag K base (halves of 16)

  v8f acc[4];
#pragma unroll
  for (int s = 0; s < 4; ++s) acc[s] = vzero8();

  // this thread's staging element pair within the 16x32 A slice
  const int e      = tid * 2;              // bf16 element index (0..511)
  const int ar_row = e >> 5;
  const int ar_col = e & 31;
  const bf16_t* Ag = A + (size_t)(mt * 16 + ar_row) * K + ar_col;
  const uint32_t lds_off = (uint32_t)(uintptr_t)(&As[e]);  // low 32 bits = LDS offset

  for (int k0 = 0; k0 < K; k0 += 32) {
    // async stage: each lane copies 4 bytes of the A slice straight into LDS
    {
      const bf16_t* g = Ag + k0;
      asm volatile("global_load_async_to_lds_b32 %0, %1, off"
                   :: "v"(lds_off), "v"(g) : "memory");
      asm volatile("s_wait_asynccnt 0x0" ::: "memory");
    }
    __syncthreads();

    v16bf a;
#pragma unroll
    for (int j = 0; j < 8; ++j) {
      a[j]     = As[ln * 32 + bkoff + j];
      a[8 + j] = As[ln * 32 + bkoff + 8 + j];
    }
#pragma unroll
    for (int sub = 0; sub < 4; ++sub) {
      v16bf bfr;
      const bf16_t* br = Bt + (size_t)(n0 + sub * 16 + ln) * K + k0 + bkoff;
      __builtin_prefetch(br + 32, 0, 1);
#pragma unroll
      for (int j = 0; j < 16; ++j) bfr[j] = br[j];   // contiguous 32B per lane
      acc[sub] = wmma_bf16(a, bfr, acc[sub]);
    }
    __syncthreads();                        // protect As before next overwrite
  }

#pragma unroll
  for (int sub = 0; sub < 4; ++sub)
#pragma unroll
    for (int r = 0; r < 8; ++r) {
      const int row = mt * 16 + r + 8 * hf;
      const int col = n0 + sub * 16 + ln;
      if (F32OUT) ((float*)Cc)[(size_t)row * N + col] = acc[sub][r];
      else        ((bf16_t*)Cc)[(size_t)row * N + col] = (bf16_t)acc[sub][r];
    }
}

// ---------------------------------------------------------------- flash attention
// One wave handles one (b, h, 16-query tile). Sliding-window causal mask.
// Q is pre-scaled by 1/sqrt(HD) (folded into RoPE). V is pre-transposed.
__global__ __launch_bounds__(128) void attn_wmma_kernel(const bf16_t* __restrict__ Q,
                                                        const bf16_t* __restrict__ K,
                                                        const bf16_t* __restrict__ Vt,
                                                        bf16_t* __restrict__ Y) {
  __shared__ bf16_t plds[4][16 * 32];      // per-wave P staging (C-layout -> A-layout)

  const int lane = threadIdx.x & 31;
  // wave coordinates made provably wave-uniform -> SALU loop control, saddr loads,
  // and EXEC provably all-ones around every WMMA.
  const int wv   = __builtin_amdgcn_readfirstlane((int)threadIdx.x >> 5);
  const int gw   = (int)blockIdx.x * 4 + wv;
  const int qt = gw & 127;                 // T/16 = 128 query tiles
  const int h  = (gw >> 7) & 15;
  const int b  = gw >> 11;

  const int hf    = lane >> 4;
  const int ln    = lane & 15;
  const int akoff = hf * 8;
  const int bkoff = hf * 16;

  const bf16_t* Qp  = Q  + (size_t)b * kT * kQC + (size_t)h * kHD;         // stride kQC
  const bf16_t* Kp  = K  + (size_t)b * kT * kKC + (size_t)(h >> 2) * kHD;  // stride kKC
  const bf16_t* Vtp = Vt + (size_t)(b * 4 + (h >> 2)) * kHD * kT;          // [128][T]

  const int qrow0 = qt * 16;
  v16bf aq[4];
#pragma unroll
  for (int kk = 0; kk < 4; ++kk) {
    const bf16_t* qr = Qp + (size_t)(qrow0 + ln) * kQC + kk * 32 + akoff;
#pragma unroll
    for (int j = 0; j < 8; ++j) { aq[kk][j] = qr[j]; aq[kk][8 + j] = qr[16 + j]; }
  }

  v16bf onesb;                             // B-frag of all-ones: P x ones = row sums
#pragma unroll
  for (int j = 0; j < 16; ++j) onesb[j] = (bf16_t)1.0f;

  v8f o[8];
#pragma unroll
  for (int i = 0; i < 8; ++i) o[i] = vzero8();
  float rm[8], rs[8];
#pragma unroll
  for (int r = 0; r < 8; ++r) { rm[r] = -__builtin_inff(); rs[r] = 0.0f; }

  const int qhi = qrow0 + 15;
  int jmin = qrow0 - (kWIN - 1); if (jmin < 0) jmin = 0;

  for (int kb = (jmin & ~31); kb <= qhi; kb += 32) {
    // ---- scores: two 16x16 tiles over the 32-key block
    v8f s0 = vzero8(), s1 = vzero8();
#pragma unroll
    for (int kk = 0; kk < 4; ++kk) {
      v16bf b0, b1;
      const bf16_t* kr0 = Kp + (size_t)(kb + ln) * kKC + kk * 32 + akoff;
      const bf16_t* kr1 = Kp + (size_t)(kb + 16 + ln) * kKC + kk * 32 + akoff;
#pragma unroll
      for (int j = 0; j < 8; ++j) {
        b0[j] = kr0[j]; b0[8 + j] = kr0[16 + j];
        b1[j] = kr1[j]; b1[8 + j] = kr1[16 + j];
      }
      s0 = wmma_bf16(aq[kk], b0, s0);
      s1 = wmma_bf16(aq[kk], b1, s1);
    }

    // ---- online softmax (max via lane shuffles; sums via WMMA below)
    float p0[8], p1[8], facs[8];
    auto softmax_rows = [&](bool masked) {
#pragma unroll
      for (int r = 0; r < 8; ++r) {
        const int m = r + 8 * hf;
        const int i = qrow0 + m;
        float v0 = s0[r];
        float v1 = s1[r];
        if (masked) {
          const int j0 = kb + ln;
          const int j1 = kb + 16 + ln;
          v0 = (j0 <= i && (i - j0) < kWIN) ? v0 : -__builtin_inff();
          v1 = (j1 <= i && (i - j1) < kWIN) ? v1 : -__builtin_inff();
        }
        float t = fmaxf(v0, v1);
        t = fmaxf(t, __shfl_xor(t, 1, 16));
        t = fmaxf(t, __shfl_xor(t, 2, 16));
        t = fmaxf(t, __shfl_xor(t, 4, 16));
        t = fmaxf(t, __shfl_xor(t, 8, 16));
        const float mnew  = fmaxf(rm[r], t);
        const float msafe = (mnew > -1e37f) ? mnew : 0.0f;  // fully-masked-row guard
        facs[r] = __expf(rm[r] - msafe);                    // exp(-inf)=0 when rm=-inf
        p0[r]   = __expf(v0 - msafe);
        p1[r]   = __expf(v1 - msafe);
        rm[r]   = mnew;
      }
    };
    // wave-uniform split: interior blocks skip the mask cndmask cascade entirely
    const bool interior = (kb + 31 <= qrow0) && (kb >= qrow0 - (kWIN - 16));
    if (interior) softmax_rows(false);
    else          softmax_rows(true);

    // rescale running output
#pragma unroll
    for (int dt = 0; dt < 8; ++dt)
#pragma unroll
      for (int r = 0; r < 8; ++r) o[dt][r] *= facs[r];

    // ---- P: C-layout -> LDS row-major -> A-layout fragment
    bf16_t* pw = &plds[wv][0];
#pragma unroll
    for (int r = 0; r < 8; ++r) {
      const int m = r + 8 * hf;
      pw[m * 32 + ln]      = (bf16_t)p0[r];
      pw[m * 32 + 16 + ln] = (bf16_t)p1[r];
    }
    asm volatile("s_wait_dscnt 0x0" ::: "memory");
    v16bf pa;
#pragma unroll
    for (int j = 0; j < 8; ++j) {
      pa[j]     = pw[ln * 32 + bkoff + j];
      pa[8 + j] = pw[ln * 32 + bkoff + 8 + j];
    }

    // ---- row sums via WMMA: P(16x32) x ones(32x16) -> each lane's c[r] = rowsum
    v8f ssum = wmma_bf16(pa, onesb, vzero8());
#pragma unroll
    for (int r = 0; r < 8; ++r) rs[r] = rs[r] * facs[r] + ssum[r];

    // ---- O += P * V  (V transposed: per-lane keys are contiguous)
#pragma unroll
    for (int dt = 0; dt < 8; ++dt) {
      v16bf bv;
      const bf16_t* vr = Vtp + (size_t)(dt * 16 + ln) * kT + kb + bkoff;
#pragma unroll
      for (int j = 0; j < 16; ++j) bv[j] = vr[j];
      o[dt] = wmma_bf16(pa, bv, o[dt]);
    }
  }

  // ---- finalize: divide by row sums, store bf16 Y[b,t,h*128+d]
#pragma unroll
  for (int r = 0; r < 8; ++r) rs[r] = 1.0f / rs[r];
  bf16_t* Yp = Y + (size_t)b * kT * kQC + (size_t)h * kHD;
#pragma unroll
  for (int dt = 0; dt < 8; ++dt)
#pragma unroll
    for (int r = 0; r < 8; ++r) {
      const int m = r + 8 * hf;
      Yp[(size_t)(qrow0 + m) * kQC + dt * 16 + ln] = (bf16_t)(o[dt][r] * rs[r]);
    }
}

// ---------------------------------------------------------------- launcher
extern "C" void kernel_launch(void* const* d_in, const int* in_sizes, int n_in,
                              void* d_out, int out_size, void* d_ws, size_t ws_size,
                              hipStream_t stream) {
  const float* x  = (const float*)d_in[0];
  const float* Wq = (const float*)d_in[1];
  const float* Wk = (const float*)d_in[2];
  const float* Wv = (const float*)d_in[3];
  const float* Wo = (const float*)d_in[4];

  bf16_t* ws = (bf16_t*)d_ws;
  size_t off = 0;
  bf16_t* xb  = ws + off; off += (size_t)kRows * 2048;   // x bf16
  bf16_t* Wqt = ws + off; off += (size_t)2048 * 2048;    // Wq^T
  bf16_t* Wkt = ws + off; off += (size_t)2048 * 512;     // Wk^T
  bf16_t* Wvt = ws + off; off += (size_t)2048 * 512;     // Wv^T
  bf16_t* Wot = ws + off; off += (size_t)2048 * 2048;    // Wo^T
  bf16_t* Qb  = ws + off; off += (size_t)kRows * kQC;
  bf16_t* Kb  = ws + off; off += (size_t)kRows * kKC;
  bf16_t* Vb  = ws + off; off += (size_t)kRows * kKC;
  bf16_t* Vtb = ws + off; off += (size_t)kRows * kKC;    // V transposed
  bf16_t* Yb  = ws + off; off += (size_t)kRows * kQC;

  // 1) fp32 -> bf16 (x) and fp32 -> bf16 transposed (weights)
  cvt_f32_bf16<<<2048, 256, 0, stream>>>(x, xb, (size_t)kRows * 2048);
  cvt_transpose<<<dim3(64, 64), 256, 0, stream>>>(Wq, Wqt, 2048, 2048);
  cvt_transpose<<<dim3(16, 64), 256, 0, stream>>>(Wk, Wkt, 2048, 512);
  cvt_transpose<<<dim3(16, 64), 256, 0, stream>>>(Wv, Wvt, 2048, 512);
  cvt_transpose<<<dim3(64, 64), 256, 0, stream>>>(Wo, Wot, 2048, 2048);

  // 2) QKV projections: blocks = (M/16) * (N/512)
  gemm_bf16_wmma<false><<<(kRows / 16) * (2048 / 512), 256, 0, stream>>>(xb, Wqt, Qb, kRows, 2048, 2048);
  gemm_bf16_wmma<false><<<(kRows / 16) * (512  / 512), 256, 0, stream>>>(xb, Wkt, Kb, kRows, 512,  2048);
  gemm_bf16_wmma<false><<<(kRows / 16) * (512  / 512), 256, 0, stream>>>(xb, Wvt, Vb, kRows, 512,  2048);

  // 3) RoPE on Q (with 1/sqrt(d) folded in) and K; transpose V for the PV matmul
  rope_kernel<<<kRows, 256, 0, stream>>>(Qb, kQC, kScale);
  rope_kernel<<<kRows, 256, 0, stream>>>(Kb, kKC, 1.0f);
  transpose_v<<<dim3(4, 64, 8), 256, 0, stream>>>(Vb, Vtb);

  // 4) sliding-window flash attention: B*H*(T/16) waves, 4 waves/block
  attn_wmma_kernel<<<(2 * kH * (kT / 16)) / 4, 128, 0, stream>>>(Qb, Kb, Vtb, Yb);

  // 5) output projection -> f32 d_out
  gemm_bf16_wmma<true><<<(kRows / 16) * (2048 / 512), 256, 0, stream>>>(Yb, Wot, d_out, kRows, 2048, 2048);
}
